// CannyL1Loss_84404697301071
// MI455X (gfx1250) — matrix operand order repaired and checked by reading, not compiled, not run on hardware
//
#include <hip/hip_runtime.h>
#include <hip/hip_bf16.h>

// CDNA5 / gfx1250 — wave32, WMMA 16x16x32 f16 + async global->LDS staging.
typedef __attribute__((ext_vector_type(16))) _Float16 v16h;
typedef __attribute__((ext_vector_type(8)))  _Float16 v8h;
typedef __attribute__((ext_vector_type(4)))  _Float16 v4h;
typedef __attribute__((ext_vector_type(8)))  float    v8f;

#define HW (512 * 512)
#define NB_FINAL 16384

// Address-space casts for the async-to-LDS builtin (params are int* in AS1/AS3
// per the compiler diagnostic).
// AS1: flat/global share the same 64-bit address value -> inttoptr is exact.
// AS3: generic LDS pointers carry the LDS offset in the low 32 bits -> trunc.
#define AS1(p) ((int __attribute__((address_space(1)))*)(unsigned long long)(p))
#define AS3(p) ((int __attribute__((address_space(3)))*)(unsigned int)(unsigned long long)(p))

#if __has_builtin(__builtin_amdgcn_global_load_async_to_lds_b32)
#define HAVE_ASYNC_LDS 1
#endif

#if __has_builtin(__builtin_amdgcn_s_wait_asynccnt)
#define WAIT_ASYNC() __builtin_amdgcn_s_wait_asynccnt(0)
#else
#define WAIT_ASYNC() asm volatile("s_wait_asynccnt 0x0" ::: "memory")
#endif

// ---------------------------------------------------------------------------
// K1: grayH(f16) = (mean_c(target)+1)*127.5 ; absd = sum_c |input-target|
// Reads input+target exactly once (the only ~100MB HBM traffic).
// ---------------------------------------------------------------------------
__global__ __launch_bounds__(256) void k_prep(const float* __restrict__ in,
                                              const float* __restrict__ tg,
                                              _Float16* __restrict__ grayH,
                                              float* __restrict__ absd) {
  int i  = blockIdx.x * 256 + threadIdx.x;   // float4 index
  int i4 = i << 2;                            // flat pixel index (b*HW + p)
  int b  = i4 >> 18;                          // HW = 2^18
  int p4 = i4 & (HW - 1);
  size_t base = (size_t)b * 3 * HW + p4;
  float4 t0 = *(const float4*)(tg + base);
  float4 t1 = *(const float4*)(tg + base + HW);
  float4 t2 = *(const float4*)(tg + base + 2 * HW);
  float4 a0 = *(const float4*)(in + base);
  float4 a1 = *(const float4*)(in + base + HW);
  float4 a2 = *(const float4*)(in + base + 2 * HW);
  v4h g;
  g[0] = (_Float16)(((t0.x + t1.x + t2.x) * (1.f / 3.f) + 1.f) * 127.5f);
  g[1] = (_Float16)(((t0.y + t1.y + t2.y) * (1.f / 3.f) + 1.f) * 127.5f);
  g[2] = (_Float16)(((t0.z + t1.z + t2.z) * (1.f / 3.f) + 1.f) * 127.5f);
  g[3] = (_Float16)(((t0.w + t1.w + t2.w) * (1.f / 3.f) + 1.f) * 127.5f);
  float4 d;
  d.x = fabsf(a0.x - t0.x) + fabsf(a1.x - t1.x) + fabsf(a2.x - t2.x);
  d.y = fabsf(a0.y - t0.y) + fabsf(a1.y - t1.y) + fabsf(a2.y - t2.y);
  d.z = fabsf(a0.z - t0.z) + fabsf(a1.z - t1.z) + fabsf(a2.z - t2.z);
  d.w = fabsf(a0.w - t0.w) + fabsf(a1.w - t1.w) + fabsf(a2.w - t2.w);
  *(v4h*)(grayH + i4) = g;
  *(float4*)(absd + i4) = d;
}

// ---------------------------------------------------------------------------
// K2: separable 5x5 Gaussian blur via V_WMMA_F32_16X16X32_F16.
// 64x64 output tile per 256-thread block (8 waves). Gray tile staged into LDS
// with GLOBAL_LOAD_ASYNC_TO_LDS_B32 (ASYNCcnt / s_wait_asynccnt), LDS
// pre-zeroed so zero-padding ('SAME') and the K=20..31 WMMA pad are free.
//   pass1: V = Wv(16x20, banded const in A) x gray(20x16 slices in B)
//   pass2: Blur = Vpatch(16x20 in A) x Wh(20x16, banded const in B)
// ---------------------------------------------------------------------------
__global__ __launch_bounds__(256) void k_blur_wmma(const _Float16* __restrict__ grayH,
                                                   float* __restrict__ blur) {
  __shared__ __align__(32) _Float16 gN[80][80];  // [row(y)][col(x)] natural layout
  __shared__ __align__(32) _Float16 Vt[64][80];  // [row][col], vertical-blurred

  const int tid = threadIdx.x;
  const int img = blockIdx.x >> 6;
  const int t   = blockIdx.x & 63;
  const int ty0 = (t >> 3) << 6;
  const int tx0 = (t & 7) << 6;
  const _Float16* gimg = grayH + (size_t)img * HW;

  // Pre-zero both LDS buffers (padding + finite values under zero weights).
  for (int i = tid; i < 80 * 80; i += 256) gN[i / 80][i % 80] = (_Float16)0.f;
  for (int i = tid; i < 64 * 80; i += 256) Vt[i / 80][i % 80] = (_Float16)0.f;
  __syncthreads();

  // Stage in-bounds gray region (rows/cols [-2, 66) of the tile) into LDS.
#if defined(HAVE_ASYNC_LDS)
  for (int i = tid; i < 68 * 34; i += 256) {   // f16 pairs (4B) per copy
    int r = i / 34, c = (i % 34) * 2;
    int gy = ty0 - 2 + r, gx = tx0 - 2 + c;    // gx even: pair never straddles edge
    if ((unsigned)gy < 512u && (unsigned)gx < 512u)
      __builtin_amdgcn_global_load_async_to_lds_b32(
          AS1(gimg + gy * 512 + gx), AS3(&gN[r][c]), 0, 0);
  }
  WAIT_ASYNC();
#else
  for (int i = tid; i < 68 * 68; i += 256) {
    int r = i / 68, c = i % 68;
    int gy = ty0 - 2 + r, gx = tx0 - 2 + c;
    if ((unsigned)gy < 512u && (unsigned)gx < 512u)
      gN[r][c] = gimg[gy * 512 + gx];
  }
#endif
  __syncthreads();

  const int  lane = tid & 31;
  const int  wv   = tid >> 5;
  const int  m    = lane & 15;
  const bool hi   = lane >= 16;

  // Gaussian (size 5, std 1), normalized — matches reference in f32.
  const float g5[5] = {0.054488685f, 0.24420134f, 0.40261996f, 0.24420134f, 0.054488685f};

  // Constant banded operands (zero for K>=20 kills padded-K contributions).
  v16h aW, bW;
#pragma unroll
  for (int e = 0; e < 16; ++e) {
    int kk = e + (hi ? 16 : 0);                            // A layout K index
    int d  = kk - m;
    aW[e] = (_Float16)((kk < 20 && d >= 0 && d < 5) ? g5[d] : 0.f);
    int K  = e + (e < 8 ? 0 : 8) + (hi ? 8 : 0);          // B layout K index
    int d2 = K - m;
    bW[e] = (_Float16)((K < 20 && d2 >= 0 && d2 < 5) ? g5[d2] : 0.f);
  }

  // Pass 1 (vertical): 4 row-subtiles x 5 col-subtiles covering 64 x 68 of V.
  for (int s = wv; s < 20; s += 8) {
    int r0 = (s / 5) << 4;
    int sc = s % 5;
    int c0 = (sc < 4) ? (sc << 4) : 52;                    // last subtile overlaps
    int rb = r0 + (hi ? 16 : 0);
    v16h bg;                                               // B: elem e -> K=e(+16 hi)
#pragma unroll
    for (int e = 0; e < 16; ++e) bg[e] = gN[rb + e][c0 + m];
    v8f acc = {};
    acc = __builtin_amdgcn_wmma_f32_16x16x32_f16(false, aW, false, bg,
                                                 (short)0, acc, false, false);
#pragma unroll
    for (int r = 0; r < 8; ++r)
      Vt[r0 + r + (hi ? 8 : 0)][c0 + m] = (_Float16)acc[r];
  }
  __syncthreads();

  // Pass 2 (horizontal): 4x4 output subtiles -> global blur.
  float* bimg = blur + (size_t)img * HW;
  for (int s = wv; s < 16; s += 8) {
    int r0 = (s >> 2) << 4;
    int c0 = (s & 3) << 4;
    v8h lo  = *(const v8h*)&Vt[r0 + m][c0 + (hi ? 8 : 0)];
    v8h hi8 = *(const v8h*)&Vt[r0 + m][c0 + 16 + (hi ? 8 : 0)];
    v16h a2;
#pragma unroll
    for (int e = 0; e < 8; ++e) { a2[e] = lo[e]; a2[e + 8] = hi8[e]; }
    v8f acc = {};
    acc = __builtin_amdgcn_wmma_f32_16x16x32_f16(false, a2, false, bW,
                                                 (short)0, acc, false, false);
    int row0 = ty0 + r0 + (hi ? 8 : 0);
    int col  = tx0 + c0 + m;
#pragma unroll
    for (int r = 0; r < 8; ++r)
      bimg[(size_t)(row0 + r) * 512 + col] = acc[r];
  }
}

// ---------------------------------------------------------------------------
// K3: Sobel (cross-correlation, zero-pad SAME) -> magnitude + direction class.
// Tile staged via async global->LDS copies as well.
// ---------------------------------------------------------------------------
__global__ __launch_bounds__(256) void k_sobel(const float* __restrict__ blur,
                                               float* __restrict__ mag,
                                               unsigned char* __restrict__ dir) {
  __shared__ float sb[18][20];
  int img = blockIdx.z;
  int by = blockIdx.y * 16, bx = blockIdx.x * 16;
  const float* bimg = blur + (size_t)img * HW;
  int tid = threadIdx.y * 16 + threadIdx.x;

  for (int i = tid; i < 18 * 20; i += 256) sb[i / 20][i % 20] = 0.f;
  __syncthreads();
#if defined(HAVE_ASYNC_LDS)
  for (int i = tid; i < 18 * 18; i += 256) {
    int r = i / 18, c = i % 18;
    int gy = by - 1 + r, gx = bx - 1 + c;
    if ((unsigned)gy < 512u && (unsigned)gx < 512u)
      __builtin_amdgcn_global_load_async_to_lds_b32(
          AS1(bimg + gy * 512 + gx), AS3(&sb[r][c]), 0, 0);
  }
  WAIT_ASYNC();
#else
  for (int i = tid; i < 18 * 18; i += 256) {
    int r = i / 18, c = i % 18;
    int gy = by - 1 + r, gx = bx - 1 + c;
    if ((unsigned)gy < 512u && (unsigned)gx < 512u) sb[r][c] = bimg[gy * 512 + gx];
  }
#endif
  __syncthreads();

  int r = threadIdx.y + 1, c = threadIdx.x + 1;
  float tl = sb[r - 1][c - 1], tm = sb[r - 1][c], tr = sb[r - 1][c + 1];
  float ml = sb[r][c - 1],                        mr = sb[r][c + 1];
  float bl = sb[r + 1][c - 1], bm = sb[r + 1][c], br = sb[r + 1][c + 1];
  float gxv = (tr - tl) + 2.f * (mr - ml) + (br - bl);
  float gyv = (bl + 2.f * bm + br) - (tl + 2.f * tm + tr);
  float mg  = sqrtf(gxv * gxv + gyv * gyv + 1e-12f);
  // direction mod 180 via |gy|/|gx| and sign(gx*gy): 0,45,90,135 -> 0,1,2,3
  float ax = fabsf(gxv), ay = fabsf(gyv);
  unsigned char d;
  if (ay >= ax * 2.4142137f)      d = 2;
  else if (ay < ax * 0.41421357f) d = 0;
  else                            d = (gxv * gyv >= 0.f) ? 1 : 3;
  size_t o = (size_t)img * HW + (size_t)(by + threadIdx.y) * 512 + (bx + threadIdx.x);
  mag[o] = mg;
  dir[o] = d;
}

// ---------------------------------------------------------------------------
// K4: non-max suppression + double threshold -> code {0,1=weak,2=strong}
// ---------------------------------------------------------------------------
__global__ __launch_bounds__(256) void k_nms(const float* __restrict__ mag,
                                             const unsigned char* __restrict__ dir,
                                             unsigned char* __restrict__ code) {
  int i = blockIdx.x * 256 + threadIdx.x;
  int b = i >> 18;
  int p = i & (HW - 1);
  int y = p >> 9, x = p & 511;
  const float* mimg = mag + (size_t)b * HW;
  float mc = mimg[p];
  int d = dir[i];
  int dy = (d == 0) ? 0 : -1;
  int dx = (d == 2) ? 0 : ((d == 3) ? -1 : 1);
  int yp = y + dy, xp = x + dx, yn = y - dy, xn = x - dx;
  float npv = ((unsigned)yp < 512u && (unsigned)xp < 512u) ? mimg[yp * 512 + xp] : 0.f;
  float nnv = ((unsigned)yn < 512u && (unsigned)xn < 512u) ? mimg[yn * 512 + xn] : 0.f;
  float nms = (mc >= npv && mc >= nnv) ? mc : 0.f;
  unsigned char cd = 0;
  if (nms > 100.f)      cd = 2;   // strong  (> th2)
  else if (nms >= 10.f) cd = 1;   // weak    (th1 <= nms <= th2)
  code[i] = cd;
}

// ---------------------------------------------------------------------------
// K5: hysteresis (3x3 strong dilation) + weighted L1 partial sums per block.
// ---------------------------------------------------------------------------
__global__ __launch_bounds__(256) void k_final(const unsigned char* __restrict__ code,
                                               const float* __restrict__ absd,
                                               float* __restrict__ partials) {
  int i = blockIdx.x * 256 + threadIdx.x;
  int b = i >> 18;
  int p = i & (HW - 1);
  int y = p >> 9, x = p & 511;
  int c = code[i];
  float edge = 0.f;
  if (c == 2) {
    edge = 1.f;
  } else if (c == 1) {
    const unsigned char* cimg = code + (size_t)b * HW;
    bool s = false;
#pragma unroll
    for (int dy = -1; dy <= 1; ++dy)
#pragma unroll
      for (int dx = -1; dx <= 1; ++dx) {
        int yy = y + dy, xx = x + dx;
        if ((unsigned)yy < 512u && (unsigned)xx < 512u)
          s = s || (cimg[yy * 512 + xx] == 2);
      }
    edge = s ? 1.f : 0.f;
  }
  float w   = 1.f + edge;                 // FACTOR = 1
  float num = w * absd[i];
  __shared__ float sn[256], sw[256];
  int tt = threadIdx.x;
  sn[tt] = num; sw[tt] = w;
  __syncthreads();
  for (int off = 128; off > 0; off >>= 1) {
    if (tt < off) { sn[tt] += sn[tt + off]; sw[tt] += sw[tt + off]; }
    __syncthreads();
  }
  if (tt == 0) {
    partials[2 * blockIdx.x]     = sn[0];
    partials[2 * blockIdx.x + 1] = sw[0];
  }
}

// ---------------------------------------------------------------------------
// K6: deterministic final reduce -> out[0] = sum(w*|d|) / sum(w)
// ---------------------------------------------------------------------------
__global__ __launch_bounds__(256) void k_reduce(const float* __restrict__ partials,
                                                float* __restrict__ out, int nb) {
  int tt = threadIdx.x;
  float n = 0.f, w = 0.f;
  for (int i = tt; i < nb; i += 256) { n += partials[2 * i]; w += partials[2 * i + 1]; }
  __shared__ float sn[256], sw[256];
  sn[tt] = n; sw[tt] = w;
  __syncthreads();
  for (int off = 128; off > 0; off >>= 1) {
    if (tt < off) { sn[tt] += sn[tt + off]; sw[tt] += sw[tt + off]; }
    __syncthreads();
  }
  if (tt == 0) out[0] = sn[0] / sw[0];
}

// ---------------------------------------------------------------------------
extern "C" void kernel_launch(void* const* d_in, const int* in_sizes, int n_in,
                              void* d_out, int out_size, void* d_ws, size_t ws_size,
                              hipStream_t stream) {
  const float* in = (const float*)d_in[0];   // input  [16,3,512,512] f32
  const float* tg = (const float*)d_in[1];   // target [16,3,512,512] f32
  float* out = (float*)d_out;                // scalar f32

  char* ws = (char*)d_ws;
  const size_t NPX = (size_t)16 * HW;        // 4,194,304
  // ws layout (~67.2 MB): 3 f32 maps + 1 f16 map + 2 u8 maps + partials.
  float*         blur  = (float*)(ws);
  float*         mag   = (float*)(ws + NPX * 4);
  float*         absd  = (float*)(ws + NPX * 8);
  _Float16*      grayH = (_Float16*)(ws + NPX * 12);
  unsigned char* dir   = (unsigned char*)(ws + NPX * 14);
  unsigned char* code  = (unsigned char*)(ws + NPX * 15);
  float*         part  = (float*)(ws + NPX * 16);

  k_prep<<<(int)(NPX / 4 / 256), 256, 0, stream>>>(in, tg, grayH, absd);
  k_blur_wmma<<<16 * 64, 256, 0, stream>>>(grayH, blur);
  k_sobel<<<dim3(32, 32, 16), dim3(16, 16), 0, stream>>>(blur, mag, dir);
  k_nms<<<NB_FINAL, 256, 0, stream>>>(mag, dir, code);
  k_final<<<NB_FINAL, 256, 0, stream>>>(code, absd, part);
  k_reduce<<<1, 256, 0, stream>>>(part, out, NB_FINAL);
}
